// GAT_64819646431498
// MI455X (gfx1250) — matrix operand order repaired and checked by reading, not compiled, not run on hardware
//
#include <hip/hip_runtime.h>
#include <hip/hip_bf16.h>

typedef __attribute__((ext_vector_type(16))) _Float16 v16h;
typedef __attribute__((ext_vector_type(8)))  float    v8f;

// ---------- helpers ----------
__device__ __forceinline__ unsigned f32_ord(float f) {
  unsigned b = __float_as_uint(f);
  return (b & 0x80000000u) ? ~b : (b | 0x80000000u);
}
__device__ __forceinline__ float ord_f32(unsigned u) {
  unsigned b = (u & 0x80000000u) ? (u ^ 0x80000000u) : ~u;
  return __uint_as_float(b);
}
__device__ __forceinline__ float sigm(float x) { return 1.0f / (1.0f + expf(-x)); }
__device__ __forceinline__ float lrelu(float x, float s) { return x > 0.0f ? x : s * x; }

// ---------- operand swizzle into WMMA fragment order ----------
// A: in [M,128] f32 row-major -> out [mt][kc][lane][16] f16 (mt=M/16, kc=K/32=4)
// lane: m = mt*16 + (lane&15), half = lane>>4
// elem 2v+b holds K = kc*32 + (v>>2)*16 + half*8 + (v&3)*2 + b   (ISA A 16x32 layout)
__global__ void k_aswz(const float* __restrict__ in, _Float16* __restrict__ out, int M) {
  int idx = blockIdx.x * blockDim.x + threadIdx.x;   // ((mt*4)+kc)*32 + lane
  int total = (M >> 4) << 7;
  if (idx >= total) return;
  int lane = idx & 31;
  int kc = (idx >> 5) & 3;
  int mt = idx >> 7;
  int half = lane >> 4;
  const float* src = in + (size_t)((mt << 4) + (lane & 15)) * 128 + (kc << 5);
  v16h frag;
#pragma unroll
  for (int v = 0; v < 8; ++v) {
    int ka = ((v >> 2) << 4) + (half << 3) + ((v & 3) << 1);
    frag[2 * v]     = (_Float16)src[ka];
    frag[2 * v + 1] = (_Float16)src[ka + 1];
  }
  *(v16h*)(out + (size_t)idx * 16) = frag;
}

// B: logical [K=128, Nc=512] -> out [j][kc][lane][16] f16 (j = col tile 0..31)
// lane: col = j*16 + (lane&15); elem e holds K = kc*32 + (e>>3)*16 + (lane>>4)*8 + (e&7), e=0..15
// transpose==0: in[k*512+col] (GAT W);  transpose==1: in[col*128+k] (LSTM weight^T)
__global__ void k_bswz(const float* __restrict__ in, _Float16* __restrict__ out,
                       int transpose) {
  int idx = blockIdx.x * blockDim.x + threadIdx.x;   // ((j*4)+kc)*32 + lane, total 4096
  if (idx >= 32 * 4 * 32) return;
  int lane = idx & 31;
  int kc = (idx >> 5) & 3;
  int j = idx >> 7;
  int col = (j << 4) + (lane & 15);
  int kbase = (kc << 5) + ((lane >> 4) << 3);
  v16h frag;
#pragma unroll
  for (int e = 0; e < 16; ++e) {
    int k = kbase + ((e >> 3) << 4) + (e & 7);
    float v = transpose ? in[(size_t)col * 128 + k] : in[(size_t)k * 512 + col];
    frag[e] = (_Float16)v;
  }
  *(v16h*)(out + (size_t)idx * 16) = frag;
}

// ---------- WMMA GEMM: C[M,512](f32) = Asw * Bsw (+C if acc); K=128 ----------
// 4 waves/block, one 16-row strip per wave; B staged to LDS in two 64KB phases
// via async-to-LDS engine; 128 v_wmma per wave per call.
__global__ __launch_bounds__(128)
void k_gemm_wmma(const _Float16* __restrict__ Asw, const _Float16* __restrict__ Bsw,
                 float* __restrict__ C, int M, int acc) {
  __shared__ _Float16 sB[16 * 4 * 32 * 16];   // 32768 halfs = 64KB (16 col tiles)
  const int tid  = threadIdx.x;
  const int wave = tid >> 5;
  const int lane = tid & 31;
  const int half = lane >> 4;
  const int lm   = lane & 15;
  const int Mt   = M >> 4;
  const int mt   = blockIdx.x * 4 + wave;
  const bool valid = mt < Mt;
  const int mts  = valid ? mt : 0;

  // A fragments for the 4 K-chunks: 32B contiguous per lane (2x global_load_b128 each)
  const v16h* Af = (const v16h*)Asw + ((size_t)mts << 2) * 32 + lane;
  v16h a0 = Af[0], a1 = Af[32], a2 = Af[64], a3 = Af[96];

  float* Crow = C + ((size_t)mts << 4) * 512;
  unsigned ldsbase = (unsigned)(size_t)&sB[0];

  for (int ph = 0; ph < 2; ++ph) {
    __syncthreads();   // previous phase's consumers done before restage
    const char* g = (const char*)Bsw + (size_t)ph * 65536;
    for (int i = tid; i < 4096; i += 128) {
      unsigned ldsa = ldsbase + (unsigned)i * 16u;
      unsigned long long ga = (unsigned long long)(size_t)(g + (size_t)i * 16);
      asm volatile("global_load_async_to_lds_b128 %0, %1, off"
                   :: "v"(ldsa), "v"(ga) : "memory");
    }
    asm volatile("s_wait_asynccnt 0x0" ::: "memory");
    __syncthreads();
    const v16h* sBf = (const v16h*)sB;
    for (int j = 0; j < 16; ++j) {
      int col0 = (((ph << 4) + j) << 4);
      v8f cacc = {};
      if (acc && valid) {
#pragma unroll
        for (int r = 0; r < 8; ++r)
          cacc[r] = Crow[(size_t)(half * 8 + r) * 512 + col0 + lm];
      }
      const v16h* bf = sBf + ((size_t)j << 2) * 32 + lane;
      cacc = __builtin_amdgcn_wmma_f32_16x16x32_f16(false, a0, false, bf[0],
                                                    (short)0, cacc, false, false);
      cacc = __builtin_amdgcn_wmma_f32_16x16x32_f16(false, a1, false, bf[32],
                                                    (short)0, cacc, false, false);
      cacc = __builtin_amdgcn_wmma_f32_16x16x32_f16(false, a2, false, bf[64],
                                                    (short)0, cacc, false, false);
      cacc = __builtin_amdgcn_wmma_f32_16x16x32_f16(false, a3, false, bf[96],
                                                    (short)0, cacc, false, false);
      if (valid) {
#pragma unroll
        for (int r = 0; r < 8; ++r)
          Crow[(size_t)(half * 8 + r) * 512 + col0 + lm] = cacc[r];
      }
    }
  }
}

// ---------- GAT kernels ----------
__global__ void k_node_att(const float* __restrict__ xp, const float* __restrict__ asrc,
                           const float* __restrict__ adst, float* __restrict__ als,
                           float* __restrict__ ald, int N) {
  int idx = blockIdx.x * blockDim.x + threadIdx.x;
  if (idx >= N * 4) return;
  int n = idx >> 2, h = idx & 3;
  const float* xr = xp + (size_t)n * 512 + h * 128;
  float ss = 0.0f, sd = 0.0f;
  for (int cc = 0; cc < 128; ++cc) {
    float v = xr[cc];
    ss += v * asrc[h * 128 + cc];
    sd += v * adst[h * 128 + cc];
  }
  als[idx] = ss;
  ald[idx] = sd;
}

__global__ void k_initmax(unsigned* m, int n) {
  int i = blockIdx.x * blockDim.x + threadIdx.x;
  if (i < n) m[i] = 0x007FFFFFu;  // ordered encoding of -inf
}

__global__ void k_edge_lmax(const int* __restrict__ src, const int* __restrict__ dst,
                            int Et, int Ne, const float* __restrict__ als,
                            const float* __restrict__ ald, float* __restrict__ elog,
                            unsigned* __restrict__ mval) {
  int e = blockIdx.x * blockDim.x + threadIdx.x;
  if (e >= Ne) return;
  int s, d;
  if (e < Et) { s = src[e]; d = dst[e]; } else { s = d = e - Et; }
  for (int h = 0; h < 4; ++h) {
    float lg = lrelu(als[s * 4 + h] + ald[d * 4 + h], 0.2f);
    elog[(size_t)e * 4 + h] = lg;
    atomicMax(&mval[d * 4 + h], f32_ord(lg));
  }
}

__global__ void k_edge_expsum(const int* __restrict__ src, const int* __restrict__ dst,
                              int Et, int Ne, float* __restrict__ elog,
                              const unsigned* __restrict__ mval, float* __restrict__ denom) {
  int e = blockIdx.x * blockDim.x + threadIdx.x;
  if (e >= Ne) return;
  int d = (e < Et) ? dst[e] : e - Et;
  for (int h = 0; h < 4; ++h) {
    float ev = expf(elog[(size_t)e * 4 + h] - ord_f32(mval[d * 4 + h]));
    elog[(size_t)e * 4 + h] = ev;
    atomicAdd(&denom[d * 4 + h], ev);
  }
}

__global__ __launch_bounds__(128)
void k_edge_msg(const int* __restrict__ src, const int* __restrict__ dst, int Et,
                const float* __restrict__ elog, const float* __restrict__ denom,
                const float* __restrict__ xp, float* __restrict__ aggr) {
  int e = blockIdx.x;
  int cc = threadIdx.x;
  int s, d;
  if (e < Et) { s = src[e]; d = dst[e]; } else { s = d = e - Et; }
  for (int h = 0; h < 4; ++h) {
    float alpha = elog[(size_t)e * 4 + h] / denom[d * 4 + h];
    atomicAdd(&aggr[(size_t)d * 512 + h * 128 + cc],
              alpha * xp[(size_t)s * 512 + h * 128 + cc]);
  }
}

__global__ void k_gat_fin(const float* __restrict__ aggr, const float* __restrict__ bias,
                          float* __restrict__ acc, int NC) {
  int idx = blockIdx.x * blockDim.x + threadIdx.x;
  if (idx >= NC) return;
  int n = idx >> 7, cc = idx & 127;
  const float* ar = aggr + (size_t)n * 512;
  float v = 0.25f * (ar[cc] + ar[128 + cc] + ar[256 + cc] + ar[384 + cc]) + bias[cc];
  acc[idx] += lrelu(v, 0.01f);
}

__global__ void k_scale(const float* __restrict__ a, float* __restrict__ o, float s, int n) {
  int i = blockIdx.x * blockDim.x + threadIdx.x;
  if (i < n) o[i] = a[i] * s;
}

// ---------- LSTM cell (PyTorch gate order i,f,g,o) ----------
__global__ void k_lstm_cell(const float* __restrict__ gates, const float* __restrict__ bih,
                            const float* __restrict__ bhh, float* __restrict__ h,
                            float* __restrict__ c, float* __restrict__ hout, int NC) {
  int idx = blockIdx.x * blockDim.x + threadIdx.x;
  if (idx >= NC) return;
  int n = idx >> 7, j = idx & 127;
  const float* g = gates + (size_t)n * 512;
  float gi = g[j]       + bih[j]       + bhh[j];
  float gf = g[128 + j] + bih[128 + j] + bhh[128 + j];
  float gg = g[256 + j] + bih[256 + j] + bhh[256 + j];
  float go = g[384 + j] + bih[384 + j] + bhh[384 + j];
  float cv = sigm(gf) * c[idx] + sigm(gi) * tanhf(gg);
  float hv = sigm(go) * tanhf(cv);
  c[idx] = cv;
  h[idx] = hv;
  hout[idx] = hv;
}

// ---------- JK attention over L=3 ----------
__global__ void k_jk(const float* __restrict__ hf, const float* __restrict__ hb,
                     const float* __restrict__ xs, const float* __restrict__ jw,
                     const float* __restrict__ jb, float* __restrict__ xjk, int N) {
  int n = blockIdx.x * blockDim.x + threadIdx.x;
  if (n >= N) return;
  size_t NC = (size_t)N * 128;
  float sc[3], mx = -1e30f;
  for (int l = 0; l < 3; ++l) {
    const float* hfr = hf + l * NC + (size_t)n * 128;
    const float* hbr = hb + l * NC + (size_t)n * 128;
    float s = jb[0];
    for (int k = 0; k < 128; ++k) s += hfr[k] * jw[k] + hbr[k] * jw[128 + k];
    sc[l] = s;
    mx = fmaxf(mx, s);
  }
  float a[3], den = 0.0f;
  for (int l = 0; l < 3; ++l) { a[l] = expf(sc[l] - mx); den += a[l]; }
  for (int l = 0; l < 3; ++l) a[l] /= den;
  for (int cc = 0; cc < 128; ++cc) {
    float v = 0.0f;
    for (int l = 0; l < 3; ++l) v += a[l] * xs[l * NC + (size_t)n * 128 + cc];
    xjk[(size_t)n * 128 + cc] = v;
  }
}

__global__ void k_pool(const float* __restrict__ xjk, const int* __restrict__ batch,
                       float* __restrict__ pooled, int NC) {
  int idx = blockIdx.x * blockDim.x + threadIdx.x;
  if (idx >= NC) return;
  int n = idx >> 7, cc = idx & 127;
  atomicAdd(&pooled[batch[n] * 128 + cc], xjk[idx]);
}

// ---------- final MLP: one block per graph ----------
__global__ __launch_bounds__(128)
void k_mlp(const float* __restrict__ pooled, const float* __restrict__ ptype,
           const float* __restrict__ w1, const float* __restrict__ b1,
           const float* __restrict__ w2, const float* __restrict__ b2,
           const float* __restrict__ w3, const float* __restrict__ b3,
           float* __restrict__ out) {
  int g = blockIdx.x;
  int t = threadIdx.x;
  __shared__ float zin[129];
  __shared__ float z1[80];
  __shared__ float z2[80];
  if (t < 128) zin[t] = pooled[g * 128 + t];
  if (t == 0) zin[128] = ptype[g];
  __syncthreads();
  if (t < 80) {
    float s = b1[t];
    for (int k = 0; k < 129; ++k) s += zin[k] * w1[k * 80 + t];
    z1[t] = lrelu(s, 0.01f);
  }
  __syncthreads();
  if (t < 80) {
    float s = b2[t];
    for (int k = 0; k < 80; ++k) s += z1[k] * w2[k * 80 + t];
    z2[t] = lrelu(s, 0.01f);
  }
  __syncthreads();
  if (t < 10) {
    float s = b3[t];
    for (int k = 0; k < 80; ++k) s += z2[k] * w3[k * 10 + t];
    out[g * 10 + t] = s;
  }
}

// ---------- host orchestration ----------
extern "C" void kernel_launch(void* const* d_in, const int* in_sizes, int n_in,
                              void* d_out, int out_size, void* d_ws, size_t ws_size,
                              hipStream_t stream) {
  const int C = 128, H = 4, T = 3, P = 2, HC = 512, NG = 64;
  const int N = in_sizes[0] / C;
  const int E = in_sizes[1] / 2;
  const int Et = E / T;
  const int Ne = Et + N;            // one edge type + self loops
  const int NC = N * C;
  const int Mt = N >> 4;            // 16-row tiles

  const float* x     = (const float*)d_in[0];
  const int*   ei    = (const int*)d_in[1];
  const int*   batch = (const int*)d_in[2];
  const float* ptype = (const float*)d_in[3];
  const float* gW    = (const float*)d_in[4];
  const float* gas   = (const float*)d_in[5];
  const float* gad   = (const float*)d_in[6];
  const float* gb    = (const float*)d_in[7];
  const float* wih[2] = {(const float*)d_in[8],  (const float*)d_in[12]};
  const float* whh[2] = {(const float*)d_in[9],  (const float*)d_in[13]};
  const float* bih[2] = {(const float*)d_in[10], (const float*)d_in[14]};
  const float* bhh[2] = {(const float*)d_in[11], (const float*)d_in[15]};
  const float* jw  = (const float*)d_in[16];
  const float* jb  = (const float*)d_in[17];
  const float* f1w = (const float*)d_in[18];
  const float* f1b = (const float*)d_in[19];
  const float* f2w = (const float*)d_in[20];
  const float* f2b = (const float*)d_in[21];
  const float* f3w = (const float*)d_in[22];
  const float* f3b = (const float*)d_in[23];
  float* out = (float*)d_out;

  // workspace carve
  char* wp = (char*)d_ws;
  auto carve = [&](size_t bytes) -> void* {
    void* r = (void*)wp;
    wp += (bytes + 255) & ~(size_t)255;
    return r;
  };
  float*    xs    = (float*)carve((size_t)3 * NC * 4);   // xs[0..2]
  float*    xp    = (float*)carve((size_t)N * HC * 4);   // GAT xp / LSTM gates
  float*    aggr  = (float*)carve((size_t)N * HC * 4);
  float*    accb  = (float*)carve((size_t)NC * 4);
  float*    hbuf  = (float*)carve((size_t)NC * 4);
  float*    cbuf  = (float*)carve((size_t)NC * 4);
  float*    hf    = (float*)carve((size_t)3 * NC * 4);
  float*    hb    = (float*)carve((size_t)3 * NC * 4);
  float*    xjk   = (float*)carve((size_t)NC * 4);
  float*    als   = (float*)carve((size_t)N * H * 4);
  float*    ald   = (float*)carve((size_t)N * H * 4);
  unsigned* mval  = (unsigned*)carve((size_t)N * H * 4);
  float*    denom = (float*)carve((size_t)N * H * 4);
  float*    elog  = (float*)carve((size_t)Ne * H * 4);
  _Float16* ah    = (_Float16*)carve((size_t)NC * 2);       // swizzled A
  _Float16* bh    = (_Float16*)carve((size_t)C * HC * 2);   // swizzled B (GAT W)
  _Float16* wih_h = (_Float16*)carve((size_t)C * HC * 2);
  _Float16* whh_h = (_Float16*)carve((size_t)C * HC * 2);
  float*    pooled = (float*)carve((size_t)NG * C * 4);

  const int TPB = 256;
  const int gNC  = (NC + TPB - 1) / TPB;
  const int gNH  = (N * H + TPB - 1) / TPB;
  const int gNe  = (Ne + TPB - 1) / TPB;
  const int gASW = (Mt * 128 + TPB - 1) / TPB;
  const int gBSW = (4096 + TPB - 1) / TPB;
  const int gemmBlocks = (Mt + 3) / 4;

  // xs[0] = x
  hipMemcpyAsync(xs, x, (size_t)NC * 4, hipMemcpyDeviceToDevice, stream);

  // ---- GAT layers ----
  for (int p = 0; p < P; ++p) {
    float* cur = xs + (size_t)p * NC;
    hipMemsetAsync(accb, 0, (size_t)NC * 4, stream);
    k_aswz<<<gASW, TPB, 0, stream>>>(cur, ah, N);
    for (int t = 0; t < T; ++t) {
      int pt = p * T + t;
      k_bswz<<<gBSW, TPB, 0, stream>>>(gW + (size_t)pt * C * HC, bh, 0);
      k_gemm_wmma<<<gemmBlocks, 128, 0, stream>>>(ah, bh, xp, N, 0);
      k_node_att<<<gNH, TPB, 0, stream>>>(xp, gas + (size_t)pt * H * C,
                                          gad + (size_t)pt * H * C, als, ald, N);
      k_initmax<<<gNH, TPB, 0, stream>>>(mval, N * H);
      hipMemsetAsync(denom, 0, (size_t)N * H * 4, stream);
      hipMemsetAsync(aggr, 0, (size_t)N * HC * 4, stream);
      const int* srcp = ei + (size_t)t * Et;
      const int* dstp = ei + (size_t)E + (size_t)t * Et;
      k_edge_lmax<<<gNe, TPB, 0, stream>>>(srcp, dstp, Et, Ne, als, ald, elog, mval);
      k_edge_expsum<<<gNe, TPB, 0, stream>>>(srcp, dstp, Et, Ne, elog, mval, denom);
      k_edge_msg<<<Ne, 128, 0, stream>>>(srcp, dstp, Et, elog, denom, xp, aggr);
      k_gat_fin<<<gNC, TPB, 0, stream>>>(aggr, gb + (size_t)pt * C, accb, NC);
    }
    k_scale<<<gNC, TPB, 0, stream>>>(accb, xs + (size_t)(p + 1) * NC, 1.0f / 3.0f, NC);
  }

  // ---- biLSTM over L=3 (gates buffer reuses xp) ----
  float* hdir[2] = {hf, hb};
  for (int d = 0; d < 2; ++d) {
    k_bswz<<<gBSW, TPB, 0, stream>>>(wih[d], wih_h, 1);  // [512,128] -> B = w^T swizzled
    k_bswz<<<gBSW, TPB, 0, stream>>>(whh[d], whh_h, 1);
    hipMemsetAsync(hbuf, 0, (size_t)NC * 4, stream);
    hipMemsetAsync(cbuf, 0, (size_t)NC * 4, stream);
    for (int s = 0; s < 3; ++s) {
      int l = (d == 0) ? s : 2 - s;
      k_aswz<<<gASW, TPB, 0, stream>>>(xs + (size_t)l * NC, ah, N);
      k_gemm_wmma<<<gemmBlocks, 128, 0, stream>>>(ah, wih_h, xp, N, 0);
      k_aswz<<<gASW, TPB, 0, stream>>>(hbuf, ah, N);
      k_gemm_wmma<<<gemmBlocks, 128, 0, stream>>>(ah, whh_h, xp, N, 1);
      k_lstm_cell<<<gNC, TPB, 0, stream>>>(xp, bih[d], bhh[d], hbuf, cbuf,
                                           hdir[d] + (size_t)l * NC, NC);
    }
  }

  // ---- JK attention, pooling, MLP ----
  k_jk<<<(N + 127) / 128, 128, 0, stream>>>(hf, hb, xs, jw, jb, xjk, N);
  hipMemsetAsync(pooled, 0, (size_t)NG * C * 4, stream);
  k_pool<<<gNC, TPB, 0, stream>>>(xjk, batch, pooled, NC);
  k_mlp<<<NG, 128, 0, stream>>>(pooled, ptype, f1w, f1b, f2w, f2b, f3w, f3b, out);

  (void)n_in; (void)out_size; (void)ws_size;
}
